// KAN_20658792694429
// MI455X (gfx1250) — matrix operand order repaired and checked by reading, not compiled, not run on hardware
//
#include <hip/hip_runtime.h>

// ---------------------------------------------------------------------------
// KAN harmonic layer on MI455X (gfx1250, wave32, WMMA).
//   out[b,h] = sum_{d,f} basis(x[b,d])_f * W[d,f,h] + sum_d b[d,h]
// M=16384, K=2816 (padded to 4096), N=256 f16 GEMM, f32 accumulation.
// v2: double-buffered LDS, async global->LDS B staging (ASYNCcnt),
//     XOR-swizzled conflict-free LDS layout, one barrier per K-chunk.
// ---------------------------------------------------------------------------

typedef _Float16 v16h __attribute__((ext_vector_type(16)));
typedef _Float16 v8h  __attribute__((ext_vector_type(8)));
typedef float    v8f  __attribute__((ext_vector_type(8)));

#define HARM     5
#define B_ROWS   16384
#define D_DIM    256
#define F_DIM    11
#define H_DIM    256
#define KPAD     (D_DIM * 16)   // 4096: each feature d padded 11 -> 16 K slots
#define NCHUNK   (KPAD / 32)    // 128 K-chunks of 32 (exactly 2 d's per chunk)
#define BLOCK_M  128

// LDS rows are exactly 32 halves (64B); 16B blocks are XOR-swizzled so that
// every b128 fragment access is bank-conflict-free across 16 lanes.
__device__ __forceinline__ int swz(int row, int blk) {
    return blk ^ (row & 3) ^ ((row >> 2) & 3);
}

__device__ __forceinline__ unsigned lds_off(const void* p) {
    // generic "shared" pointers carry the raw LDS offset in their low 32 bits
    return (unsigned)(uintptr_t)p;
}

// --------------------------------------------------------------------------
// Kernel 1: W[D][11][H] f32  ->  WhT[N=H][K=KPAD] f16 (transposed, padded)
// --------------------------------------------------------------------------
__global__ __launch_bounds__(256) void kan_convert_w(const float* __restrict__ W,
                                                     _Float16* __restrict__ WhT) {
    const int d = blockIdx.x;     // 0..255
    const int n = threadIdx.x;    // 0..255  (output column h)
    v8h lo, hi;
#pragma unroll
    for (int f = 0; f < 8; ++f)
        lo[f] = (_Float16)W[((size_t)(d * F_DIM + f)) * H_DIM + n];
#pragma unroll
    for (int f = 8; f < 16; ++f)
        hi[f - 8] = (f < F_DIM) ? (_Float16)W[((size_t)(d * F_DIM + f)) * H_DIM + n]
                                : (_Float16)0.0f;
    v8h* dst = (v8h*)(WhT + (size_t)n * KPAD + d * 16);   // 32B aligned
    dst[0] = lo;
    dst[1] = hi;
}

// --------------------------------------------------------------------------
// Kernel 2: sumb[h] = sum_d b[d][h]
// --------------------------------------------------------------------------
__global__ __launch_bounds__(256) void kan_bias_sum(const float* __restrict__ b,
                                                    float* __restrict__ sumb) {
    __shared__ float red[256];
    const int hl = threadIdx.x & 31;
    const int dg = threadIdx.x >> 5;
    const int h  = blockIdx.x * 32 + hl;
    float a = 0.0f;
    for (int d = dg; d < D_DIM; d += 8)
        a += b[(size_t)d * H_DIM + h];
    red[threadIdx.x] = a;
    __syncthreads();
    if (dg == 0) {
        float t = 0.0f;
#pragma unroll
        for (int g = 0; g < 8; ++g) t += red[g * 32 + hl];
        sumb[h] = t;
    }
}

// --------------------------------------------------------------------------
// Kernel 3: fused basis + WMMA GEMM, double-buffered, async B staging.
// Block: 128x256 of out; 8 waves, each owns 32(M) x 128(N).
// --------------------------------------------------------------------------
__global__ __launch_bounds__(256) void kan_wmma_gemm(const float* __restrict__ x,
                                                     const _Float16* __restrict__ WhT,
                                                     const float* __restrict__ sumb,
                                                     float* __restrict__ out) {
    __shared__ _Float16 As[2][BLOCK_M * 32];   // 2 x 8 KB
    __shared__ _Float16 Bs[2][H_DIM  * 32];    // 2 x 16 KB

    const int tid   = threadIdx.x;
    const int lane  = tid & 31;
    const int wave  = tid >> 5;
    const int mgrp  = wave & 3;    // mtiles 2*mgrp, 2*mgrp+1
    const int nhalf = wave >> 2;   // ntiles nhalf*8 .. nhalf*8+7
    const int l16   = lane & 15;
    const int kh    = lane >> 4;   // lane half selects K sub-range per ISA layout

    const int row0 = blockIdx.x * BLOCK_M;

    v8f acc[2][8];
    {
        v8f z = {};
#pragma unroll
        for (int mi = 0; mi < 2; ++mi)
#pragma unroll
            for (int ni = 0; ni < 8; ++ni) acc[mi][ni] = z;
    }

    // staging assignment: thread t -> B row n=t; A (row, which-of-2-d's)
    const int arow = tid >> 1;
    const int adl  = tid & 1;
    const float* xrow = x + (size_t)(row0 + arow) * D_DIM;

    const unsigned bs_base = lds_off(&Bs[0][0]);
    const unsigned bs_row  = bs_base + (unsigned)tid * 64u;   // this thread's B row

    // ---- staging helper (chunk ck into buffer buf) ----
    auto stage = [&](int ck, int buf) {
        // B: 4 async b128 global->LDS copies (no VGPR data path, ASYNCcnt)
        const unsigned gbase = (unsigned)tid * (KPAD * 2u) + (unsigned)ck * 64u;
        const unsigned lbase = bs_row + (unsigned)buf * (unsigned)(H_DIM * 32 * 2);
#pragma unroll
        for (int jj = 0; jj < 4; ++jj) {
            const unsigned loff = lbase + (unsigned)(swz(tid, jj) * 16);
            const unsigned goff = gbase + (unsigned)(jj * 16);
            asm volatile("global_load_async_to_lds_b128 %0, %1, %2"
                         :: "v"(loff), "v"(goff), "s"(WhT)
                         : "memory");
        }
        // A: harmonic basis for d = 2*ck + adl at row arow
        const float xv = xrow[2 * ck + adl];
        const float s1 = __sinf(xv);
        const float c1 = __cosf(xv);
        float sv[HARM], cv[HARM];
        sv[0] = s1; cv[0] = c1;
#pragma unroll
        for (int k = 1; k < HARM; ++k) {   // angle-addition recurrence: FMAs co-exec with WMMA
            sv[k] = sv[k - 1] * c1 + cv[k - 1] * s1;
            cv[k] = cv[k - 1] * c1 - sv[k - 1] * s1;
        }
        v8h h0, h1;
        h0[0] = (_Float16)1.0f;
        h0[1] = (_Float16)sv[0]; h0[2] = (_Float16)cv[0];
        h0[3] = (_Float16)sv[1]; h0[4] = (_Float16)cv[1];
        h0[5] = (_Float16)sv[2]; h0[6] = (_Float16)cv[2];
        h0[7] = (_Float16)sv[3];
        h1[0] = (_Float16)cv[3];
        h1[1] = (_Float16)sv[4]; h1[2] = (_Float16)cv[4];
        h1[3] = (_Float16)0.0f; h1[4] = (_Float16)0.0f;
        h1[5] = (_Float16)0.0f; h1[6] = (_Float16)0.0f; h1[7] = (_Float16)0.0f;
        _Float16* abase = &As[buf][arow * 32];
        *(v8h*)(abase + swz(arow, 2 * adl    ) * 8) = h0;
        *(v8h*)(abase + swz(arow, 2 * adl + 1) * 8) = h1;
    };

    // ---- prologue: stage chunk 0 ----
    stage(0, 0);
    asm volatile("s_wait_asynccnt 0x0" ::: "memory");
    __syncthreads();

    for (int c = 0; c < NCHUNK; ++c) {
        const int p = c & 1;

        // kick off next chunk's staging; its latency hides under this chunk's WMMAs
        if (c + 1 < NCHUNK) stage(c + 1, p ^ 1);

        const _Float16* Ab = &As[p][0];
        const _Float16* Bb = &Bs[p][0];

        // A fragments (ISA 16-bit A 16x32: lane m=l16; K = kh*8+[0,8) U 16+kh*8+[0,8))
        v16h afr[2];
#pragma unroll
        for (int mi = 0; mi < 2; ++mi) {
            const int m = (mgrp * 2 + mi) * 16 + l16;
            const v8h alo = *(const v8h*)(Ab + m * 32 + swz(m, kh    ) * 8);
            const v8h ahi = *(const v8h*)(Ab + m * 32 + swz(m, kh + 2) * 8);
            afr[mi] = __builtin_shufflevector(alo, ahi, 0, 1, 2, 3, 4, 5, 6, 7,
                                              8, 9, 10, 11, 12, 13, 14, 15);
        }
        // B fragments (lane n=l16; K = kh*8+[0,8) U 16+kh*8+[0,8)) + 16 WMMAs
#pragma unroll
        for (int ni = 0; ni < 8; ++ni) {
            const int n = (nhalf * 8 + ni) * 16 + l16;
            const v8h blo = *(const v8h*)(Bb + n * 32 + swz(n, kh    ) * 8);
            const v8h bhi = *(const v8h*)(Bb + n * 32 + swz(n, kh + 2) * 8);
            const v16h bfr = __builtin_shufflevector(blo, bhi, 0, 1, 2, 3, 4, 5, 6, 7,
                                                     8, 9, 10, 11, 12, 13, 14, 15);
#pragma unroll
            for (int mi = 0; mi < 2; ++mi)
                acc[mi][ni] = __builtin_amdgcn_wmma_f32_16x16x32_f16(
                    /*neg_a=*/false, afr[mi], /*neg_b=*/false, bfr,
                    /*c_mod=*/(short)0, acc[mi][ni],
                    /*reuse_a=*/false, /*reuse_b=*/false);
        }

        // next buffer's async B data must be in LDS before anyone passes the barrier
        asm volatile("s_wait_asynccnt 0x0" ::: "memory");
        __syncthreads();   // also flushes this iteration's A ds_stores (dscnt)
    }

    // ---- epilogue: C/D layout (VGPR v -> M=v or v+8 by lane half; lane -> N) ----
#pragma unroll
    for (int mi = 0; mi < 2; ++mi) {
        const int mbase = row0 + (mgrp * 2 + mi) * 16 + (kh << 3);
#pragma unroll
        for (int ni = 0; ni < 8; ++ni) {
            const int ncol = (nhalf * 8 + ni) * 16 + l16;
            const float sb = sumb[ncol];
            float* op = out + (size_t)mbase * H_DIM + ncol;
#pragma unroll
            for (int v = 0; v < 8; ++v)
                op[(size_t)v * H_DIM] = acc[mi][ni][v] + sb;
        }
    }
}

// --------------------------------------------------------------------------
extern "C" void kernel_launch(void* const* d_in, const int* in_sizes, int n_in,
                              void* d_out, int out_size, void* d_ws, size_t ws_size,
                              hipStream_t stream) {
    (void)in_sizes; (void)n_in; (void)out_size; (void)ws_size;
    const float* x = (const float*)d_in[0];   // [16384, 256] f32
    const float* W = (const float*)d_in[1];   // [256, 11, 256] f32
    const float* b = (const float*)d_in[2];   // [256, 256] f32
    float* out = (float*)d_out;               // [16384, 256] f32

    _Float16* WhT = (_Float16*)d_ws;                                   // 2 MB
    float* sumb = (float*)((char*)d_ws + (size_t)H_DIM * KPAD * sizeof(_Float16));

    kan_convert_w<<<D_DIM, 256, 0, stream>>>(W, WhT);
    kan_bias_sum<<<H_DIM / 32, 256, 0, stream>>>(b, sumb);
    kan_wmma_gemm<<<B_ROWS / BLOCK_M, 256, 0, stream>>>(x, WhT, sumb, out);
}
